// Conttextual_loss_27754078667510
// MI455X (gfx1250) — matrix operand order repaired and checked by reading, not compiled, not run on hardware
//
#include <hip/hip_runtime.h>
#include <hip/hip_fp16.h>

typedef __attribute__((ext_vector_type(8)))  _Float16 v8h;
typedef __attribute__((ext_vector_type(16))) _Float16 v16h;
typedef __attribute__((ext_vector_type(8)))  float    v8f;

#define BB 4
#define CC 128
#define NN 4096
#define HPAR 0.1f
#define EPS_NORM 1e-10f
#define EPS_MIN 0.001f

// LDS tile: 16 rows x 128 f16, padded to 136 f16/row (272 B) so the
// strided per-lane ds_load_b128 B-fragment reads spread over all 64 banks.
#define YPAD 136

// ---------------- workspace layout (byte offsets) ----------------
// mean     : 512 f32        @ 0          (2048 B)
// meanpart : 4*32*128 f32   @ 2048       (65536 B)
// smax     : 4*4096 f32     @ 67584      (65536 B)
// maxA     : 4*4096 f32     @ 133120     (65536 B)
// xn       : 4*4096*128 f16 @ 200704     (4 MB)
// yn       : xn + 4 MB                     (4 MB)    total ~8.6 MB
#define WS_MEAN  0
#define WS_MPART 2048
#define WS_SMAX  67584
#define WS_MAXA  133120
#define WS_XN    200704
#define WS_YN    (WS_XN + (size_t)BB*NN*CC*2)

// ---------------------------------------------------------------
// Phase 0a: partial spatial sums of feature_y. Deterministic two-stage:
// block (b, slice) sums 128 rows into meanpart[b][slice][c].
__global__ void mean_part_kernel(const float* __restrict__ fy,
                                 float* __restrict__ mpart) {
    int b = blockIdx.x >> 5;        // 4 batches x 32 slices
    int sl = blockIdx.x & 31;
    int c = threadIdx.x;
    const float* p = fy + (size_t)(b * NN + sl * 128) * CC + c;
    float s = 0.0f;
    for (int n = 0; n < 128; ++n) s += p[(size_t)n * CC];
    mpart[(b * 32 + sl) * CC + c] = s;
}
// Phase 0b: combine 32 slices -> mean[b][c].
__global__ void mean_comb_kernel(const float* __restrict__ mpart,
                                 float* __restrict__ mean) {
    int b = blockIdx.x;
    int c = threadIdx.x;
    float s = 0.0f;
    for (int sl = 0; sl < 32; ++sl) s += mpart[(b * 32 + sl) * CC + c];
    mean[b * CC + c] = s * (1.0f / NN);
}

// ---------------------------------------------------------------
// Phase 1: center by mean_y, L2-normalize over channels, emit f16 rows [b][n][c].
__global__ void normalize_kernel(const float* __restrict__ fx,
                                 const float* __restrict__ fy,
                                 const float* __restrict__ mean,
                                 _Float16* __restrict__ xn,
                                 _Float16* __restrict__ yn) {
    int b = blockIdx.x >> 12;
    int n = blockIdx.x & (NN - 1);
    int c = threadIdx.x;
    size_t idx = ((size_t)(b * NN + n)) * CC + c;
    float m  = mean[b * CC + c];
    float vx = fx[idx] - m;
    float vy = fy[idx] - m;
    __shared__ float sx[CC];
    __shared__ float sy[CC];
    sx[c] = vx * vx;
    sy[c] = vy * vy;
    __syncthreads();
    for (int off = CC / 2; off > 0; off >>= 1) {
        if (c < off) { sx[c] += sx[c + off]; sy[c] += sy[c + off]; }
        __syncthreads();
    }
    float inx = 1.0f / (sqrtf(sx[0]) + EPS_NORM);
    float iny = 1.0f / (sqrtf(sy[0]) + EPS_NORM);
    xn[idx] = (_Float16)(vx * inx);
    yn[idx] = (_Float16)(vy * iny);
}

// ---------------------------------------------------------------
// A-fragment (16x32 f16, ISA 7.12.2): lane = row M (lane%16), lane-half selects
// K in {0..7,16..23} (half 0) or {8..15,24..31} (half 1).
__device__ __forceinline__ v16h load_fragA(const _Float16* __restrict__ rowp,
                                           int kk, int half) {
    union { v16h v; struct { v8h lo, hi; } s; } u;
    u.s.lo = *(const v8h*)(rowp + kk +      8 * half);
    u.s.hi = *(const v8h*)(rowp + kk + 16 + 8 * half);
    return u.v;
}

// ---------------------------------------------------------------
// Fused GEMM passes. One wave per 16-row tile of X (A registers live across the
// whole column sweep). All 8 waves of a block share each 16x128 Y column tile
// staged in double-buffered LDS (ds_load_b128 B-fragments), cutting Y L2
// traffic 8x vs per-wave streaming. PASS_SUM=false: row max of s = <x_n,y_m>.
// PASS_SUM=true: S[n] = sum_m exp(alpha[n]*(s-smax[n])), store maxA[n]=1/S[n].
template <bool PASS_SUM>
__global__ void gemm_pass_kernel(const _Float16* __restrict__ xn,
                                 const _Float16* __restrict__ yn,
                                 float* __restrict__ smax,
                                 float* __restrict__ maxA) {
    __shared__ __align__(16) _Float16 yt[2][16 * YPAD];

    int t    = threadIdx.x;
    int lane = t & 31;
    int wave = t >> 5;
    int b    = blockIdx.x >> 5;            // 8 tiles/block, 256 tiles/batch
    int tile = blockIdx.x * 8 + wave;
    int n0   = (tile & 255) << 4;
    int half = lane >> 4;
    int mrow = lane & 15;

    const _Float16* xb = xn + (size_t)b * NN * CC;
    const _Float16* yb = yn + (size_t)b * NN * CC;

    // Per-wave A fragments: rows n0..n0+15, all 4 k-steps (32 VGPRs), reused
    // across all 256 column tiles.
    v16h afrag[4];
    const _Float16* arow = xb + (size_t)(n0 + mrow) * CC;
#pragma unroll
    for (int k = 0; k < 4; ++k) afrag[k] = load_fragA(arow, 32 * k, half);

    // D layout: VGPR r, lanes 0-15 -> row n0+r, lanes 16-31 -> row n0+r+8.
    float acc8[8], mx8[8], al8[8];
#pragma unroll
    for (int r = 0; r < 8; ++r) {
        if (PASS_SUM) {
            float sm = smax[(size_t)b * NN + n0 + half * 8 + r];
            mx8[r]  = sm;
            al8[r]  = 1.0f / (HPAR * (1.0f - sm + EPS_MIN));
            acc8[r] = 0.0f;
        } else {
            acc8[r] = -3.0e38f;
        }
    }

    // Cooperative staging: 256 threads x 16 B = one 16x128 f16 tile.
    int srow = t >> 4;                     // 0..15
    int sseg = t & 15;                     // 0..15 (x 8 f16)
    v8h reg = *(const v8h*)(yb + (size_t)srow * CC + sseg * 8);  // tile 0

    for (int mt = 0; mt < 256; ++mt) {
        int cur = mt & 1;
        *(v8h*)(&yt[cur][srow * YPAD + sseg * 8]) = reg;
        __syncthreads();   // stores visible; also fences reuse of this buffer
        if (mt + 1 < 256)  // issue next tile's global load; consumed next iter
            reg = *(const v8h*)(yb + (size_t)((mt + 1) * 16 + srow) * CC + sseg * 8);
        if (mt + 2 < 256)
            __builtin_prefetch(yb + (size_t)((mt + 2) * 16 + srow) * CC + sseg * 8, 0, 3);

        const _Float16* yl = &yt[cur][0];
        v8f c = {};
#pragma unroll
        for (int k = 0; k < 4; ++k) {
            // B-fragment (32x16): lane = column (lane%16) = Y row in tile,
            // lane-half 0 -> K {0..7,16..23}, half 1 -> K {8..15,24..31} (matches A).
            union { v16h v; struct { v8h lo, hi; } s; } u;
            u.s.lo = *(const v8h*)(yl + mrow * YPAD + 32 * k + 8 * half);
            u.s.hi = *(const v8h*)(yl + mrow * YPAD + 32 * k + 16 + 8 * half);
            c = __builtin_amdgcn_wmma_f32_16x16x32_f16(
                    false, afrag[k], false, u.v, (short)0, c, false, false);
        }
#pragma unroll
        for (int r = 0; r < 8; ++r) {
            if (PASS_SUM) acc8[r] += __expf(al8[r] * (c[r] - mx8[r]));
            else          acc8[r] = fmaxf(acc8[r], c[r]);
        }
    }

    // Row reductions live across the 16 lanes of one half: xor-reduce.
#pragma unroll
    for (int m = 1; m < 16; m <<= 1) {
#pragma unroll
        for (int r = 0; r < 8; ++r) {
            float o = __shfl_xor(acc8[r], m, 32);
            acc8[r] = PASS_SUM ? (acc8[r] + o) : fmaxf(acc8[r], o);
        }
    }
    if (mrow == 0) {
#pragma unroll
        for (int r = 0; r < 8; ++r) {
            size_t row = (size_t)b * NN + n0 + half * 8 + r;
            if (PASS_SUM) maxA[row] = 1.0f / acc8[r];  // = max_m A[row,:]
            else          smax[row] = acc8[r];
        }
    }
}

// ---------------------------------------------------------------
// Deterministic final reduction: out[b] = -log(mean_n maxA[b,n]).
__global__ void finalize_kernel(const float* __restrict__ maxA,
                                float* __restrict__ out) {
    int b = blockIdx.x;
    int t = threadIdx.x;     // 256 threads
    float s = 0.0f;
    for (int i = t; i < NN; i += 256) s += maxA[(size_t)b * NN + i];
    __shared__ float red[256];
    red[t] = s;
    __syncthreads();
    for (int off = 128; off > 0; off >>= 1) {
        if (t < off) red[t] += red[t + off];
        __syncthreads();
    }
    if (t == 0) out[b] = -logf(red[0] * (1.0f / NN));
}

// ---------------------------------------------------------------
extern "C" void kernel_launch(void* const* d_in, const int* in_sizes, int n_in,
                              void* d_out, int out_size, void* d_ws, size_t ws_size,
                              hipStream_t stream) {
    const float* fx = (const float*)d_in[0];
    const float* fy = (const float*)d_in[1];
    float* out = (float*)d_out;

    char* wsb = (char*)d_ws;
    float*    mean  = (float*)(wsb + WS_MEAN);
    float*    mpart = (float*)(wsb + WS_MPART);
    float*    smax  = (float*)(wsb + WS_SMAX);
    float*    maxA  = (float*)(wsb + WS_MAXA);
    _Float16* xn    = (_Float16*)(wsb + WS_XN);
    _Float16* yn    = (_Float16*)(wsb + WS_YN);

    mean_part_kernel<<<BB * 32, CC, 0, stream>>>(fy, mpart);
    mean_comb_kernel<<<BB, CC, 0, stream>>>(mpart, mean);
    normalize_kernel<<<BB * NN, CC, 0, stream>>>(fx, fy, mean, xn, yn);
    gemm_pass_kernel<false><<<128, 256, 0, stream>>>(xn, yn, smax, maxA);
    gemm_pass_kernel<true ><<<128, 256, 0, stream>>>(xn, yn, smax, maxA);
    finalize_kernel<<<BB, 256, 0, stream>>>(maxA, out);
}
